// MultiheadSelectiveAttentionWithTokenPruning_21354577395913
// MI455X (gfx1250) — matrix operand order repaired and checked by reading, not compiled, not run on hardware
//
#include <hip/hip_runtime.h>
#include <math.h>

// MultiheadSelectiveAttentionWithTokenPruning for gfx1250 (MI455X, wave32, WMMA).
// B=2, N=2048, D=1024, H=16, Dh=64, BUDGET=1024, start_pos=0 (hardcoded per setup).

typedef __attribute__((ext_vector_type(16))) __bf16 v16bf;
typedef __attribute__((ext_vector_type(16))) unsigned short v16u;
typedef __attribute__((ext_vector_type(8))) float v8f;
typedef __attribute__((ext_vector_type(8))) unsigned int v8u;
typedef __attribute__((ext_vector_type(4))) unsigned int v4u;

static __device__ __forceinline__ unsigned short f2bu(float f) {
  // fp32 -> bf16 (round to nearest even)
  unsigned int x = __builtin_bit_cast(unsigned int, f);
  unsigned int r = x + 0x7FFFu + ((x >> 16) & 1u);
  return (unsigned short)(r >> 16);
}
static __device__ __forceinline__ v16bf asbf(v16u u) {
  return __builtin_bit_cast(v16bf, u);
}
static __device__ __forceinline__ v16bf asbf8(v8u u) {
  return __builtin_bit_cast(v16bf, u);
}

// ---------------------------------------------------------------------------
// GEMM: C[4096 x 1024] = A[4096 x 1024] * W[1024 x 1024], bf16 WMMA, f32 accum.
// Block: 128 threads (4 waves). Block tile 64x128; wave tile 32x64 (2x4 WMMA).
// ---------------------------------------------------------------------------
__global__ __launch_bounds__(128) void gemm_bf16_wmma(const float* __restrict__ A,
                                                      const float* __restrict__ W,
                                                      float* __restrict__ C) {
  __shared__ __align__(16) unsigned short As[64][36];    // [row][k] bf16
  __shared__ __align__(16) unsigned short Ws[128][36];   // [n][k]  bf16 (W transposed)
  const int tid = threadIdx.x;
  const int lane = tid & 31;
  const int wave = tid >> 5;
  const int wr = wave >> 1, wc = wave & 1;
  const int half = lane >> 4, l16 = lane & 15;
  const int rowBase = blockIdx.x * 64;
  const int colBase = blockIdx.y * 128;

  v8f acc[2][4];
#pragma unroll
  for (int ai = 0; ai < 2; ++ai)
#pragma unroll
    for (int t = 0; t < 4; ++t)
#pragma unroll
      for (int r = 0; r < 8; ++r) acc[ai][t][r] = 0.0f;

  for (int k0 = 0; k0 < 1024; k0 += 32) {
    __syncthreads();
    // stage A tile 64x32 (float4 loads, packed 64-bit LDS stores)
#pragma unroll
    for (int p = 0; p < 4; ++p) {
      int e4 = tid + p * 128;
      int r = e4 >> 3, c4 = (e4 & 7) * 4;
      float4 v = *(const float4*)&A[(size_t)(rowBase + r) * 1024 + k0 + c4];
      unsigned long long pk =
          (unsigned long long)((unsigned)f2bu(v.x) | ((unsigned)f2bu(v.y) << 16)) |
          ((unsigned long long)((unsigned)f2bu(v.z) | ((unsigned)f2bu(v.w) << 16)) << 32);
      *(unsigned long long*)&As[r][c4] = pk;
    }
    // stage W tile 32x128, transposed (float4 loads, scalar transposed stores)
#pragma unroll
    for (int p = 0; p < 8; ++p) {
      int e4 = tid + p * 128;
      int kk = e4 >> 5, c4 = (e4 & 31) * 4;
      float4 v = *(const float4*)&W[(size_t)(k0 + kk) * 1024 + colBase + c4];
      Ws[c4 + 0][kk] = f2bu(v.x);
      Ws[c4 + 1][kk] = f2bu(v.y);
      Ws[c4 + 2][kk] = f2bu(v.z);
      Ws[c4 + 3][kk] = f2bu(v.w);
    }
    __syncthreads();

    // A fragments: 16-bit A 16x32 layout: lane l16=row; elem e -> K = e+(e>=8?8:0)+8*half
    v16u au[2];
#pragma unroll
    for (int ai = 0; ai < 2; ++ai) {
      int row = 32 * wr + 16 * ai + l16;
#pragma unroll
      for (int i = 0; i < 8; ++i) {
        int k2 = ((i < 4) ? (2 * i) : (2 * i + 8)) + 8 * half;
        unsigned int u = *(const unsigned int*)&As[row][k2];
        au[ai][2 * i] = (unsigned short)u;
        au[ai][2 * i + 1] = (unsigned short)(u >> 16);
      }
    }
    // B fragments: 16-bit B 32x16 layout: lane l16=col; elem e -> K = e+(e>=8?8:0)+8*half
#pragma unroll
    for (int t = 0; t < 4; ++t) {
      int n = 64 * wc + 16 * t + l16;
      v16u bu;
#pragma unroll
      for (int i = 0; i < 8; ++i) {
        unsigned int u = *(const unsigned int*)&Ws[n][((i < 4) ? (2 * i) : (2 * i + 8)) + 8 * half];
        bu[2 * i] = (unsigned short)u;
        bu[2 * i + 1] = (unsigned short)(u >> 16);
      }
      v16bf bf = asbf(bu);
#pragma unroll
      for (int ai = 0; ai < 2; ++ai)
        acc[ai][t] = __builtin_amdgcn_wmma_f32_16x16x32_bf16(
            false, asbf(au[ai]), false, bf, (short)0, acc[ai][t], false, false);
    }
  }
  // D layout: lane l16=col; elem r -> row = r + 8*half
#pragma unroll
  for (int ai = 0; ai < 2; ++ai)
#pragma unroll
    for (int t = 0; t < 4; ++t)
#pragma unroll
      for (int r = 0; r < 8; ++r) {
        int row = rowBase + 32 * wr + 16 * ai + r + 8 * half;
        int col = colBase + 64 * wc + 16 * t + l16;
        C[(size_t)row * 1024 + col] = acc[ai][t][r];
      }
}

// ---------------------------------------------------------------------------
// Row LayerNorm over D=1024, in place; also emits a bf16 copy for attention.
// One block (256 thr) per row.
// ---------------------------------------------------------------------------
__global__ __launch_bounds__(256) void ln_kernel(float* __restrict__ Y,
                                                 const float* __restrict__ g,
                                                 const float* __restrict__ be,
                                                 unsigned short* __restrict__ Y16) {
  __shared__ float red[256];
  const int row = blockIdx.x;
  const int t = threadIdx.x;
  float v[4];
#pragma unroll
  for (int i = 0; i < 4; ++i) v[i] = Y[(size_t)row * 1024 + t + 256 * i];
  red[t] = v[0] + v[1] + v[2] + v[3];
  __syncthreads();
  for (int o = 128; o > 0; o >>= 1) {
    if (t < o) red[t] += red[t + o];
    __syncthreads();
  }
  const float mu = red[0] * (1.0f / 1024.0f);
  __syncthreads();
  float d = 0.0f;
#pragma unroll
  for (int i = 0; i < 4; ++i) { float x = v[i] - mu; d += x * x; }
  red[t] = d;
  __syncthreads();
  for (int o = 128; o > 0; o >>= 1) {
    if (t < o) red[t] += red[t + o];
    __syncthreads();
  }
  const float inv = rsqrtf(red[0] * (1.0f / 1024.0f) + 1e-5f);
#pragma unroll
  for (int i = 0; i < 4; ++i) {
    int c = t + 256 * i;
    float r = (v[i] - mu) * inv * g[c] + be[c];
    Y[(size_t)row * 1024 + c] = r;
    Y16[(size_t)row * 1024 + c] = f2bu(r);
  }
}

// ---------------------------------------------------------------------------
// V transpose + bf16: Vt[(b*16+h)*64 + dh][n] = bf16(V[b][n][h*64+dh]).
// Block 256 thr; 64(n) x 64(dh) tile per (b,h).
// ---------------------------------------------------------------------------
__global__ __launch_bounds__(256) void vtrans_kernel(const float* __restrict__ Vb,
                                                     unsigned short* __restrict__ Vt) {
  __shared__ float tile[64][65];
  const int bh = blockIdx.x;   // 0..31 = b*16+h
  const int nt = blockIdx.y;   // 0..31
  const int b = bh >> 4, h = bh & 15;
  const int t = threadIdx.x;
#pragma unroll
  for (int p = 0; p < 16; ++p) {
    int e = t + p * 256; int r = e >> 6, c = e & 63;   // r = n, c = dh
    tile[r][c] = Vb[((size_t)(b * 2048 + nt * 64 + r)) * 1024 + h * 64 + c];
  }
  __syncthreads();
#pragma unroll
  for (int p = 0; p < 16; ++p) {
    int e = t + p * 256; int r = e >> 6, c = e & 63;   // r = dh, c = n
    Vt[((size_t)(bh * 64 + r)) * 2048 + nt * 64 + c] = f2bu(tile[c][r]);
  }
}

// ---------------------------------------------------------------------------
// Head-0 eviction scores: S[b][j][m] = (0<m<j) ? relu(q0_j . k0_m / 8) : 0.
// ---------------------------------------------------------------------------
__global__ __launch_bounds__(256) void s0_kernel(const float* __restrict__ Q,
                                                 const float* __restrict__ K,
                                                 float* __restrict__ S) {
  __shared__ float Qs[32][64];
  __shared__ float Ks[64][65];
  const int mt = blockIdx.x, jt = blockIdx.y, b = blockIdx.z;
  const int t = threadIdx.x;
  const int m_loc = t & 63, jgrp = t >> 6;

  if (mt * 64 >= jt * 32 + 31) {   // all-zero tile (no m < j in range)
#pragma unroll
    for (int jj = 0; jj < 8; ++jj) {
      int j = jt * 32 + jgrp * 8 + jj;
      S[((size_t)(b * 2048 + j)) * 2048 + mt * 64 + m_loc] = 0.0f;
    }
    return;
  }
#pragma unroll
  for (int p = 0; p < 8; ++p) {
    int e = t + p * 256; int r = e >> 6, k = e & 63;
    Qs[r][k] = Q[((size_t)(b * 2048 + jt * 32 + r)) * 1024 + k];   // head 0 cols
  }
#pragma unroll
  for (int p = 0; p < 16; ++p) {
    int e = t + p * 256; int r = e >> 6, k = e & 63;
    Ks[r][k] = K[((size_t)(b * 2048 + mt * 64 + r)) * 1024 + k];
  }
  __syncthreads();
  float accv[8];
#pragma unroll
  for (int i = 0; i < 8; ++i) accv[i] = 0.0f;
  for (int k = 0; k < 64; ++k) {
    float kv = Ks[m_loc][k];
#pragma unroll
    for (int jj = 0; jj < 8; ++jj) accv[jj] += Qs[jgrp * 8 + jj][k] * kv;
  }
  const int m = mt * 64 + m_loc;
#pragma unroll
  for (int jj = 0; jj < 8; ++jj) {
    int j = jt * 32 + jgrp * 8 + jj;
    float val = ((m > 0) && (m < j)) ? fmaxf(accv[jj] * 0.125f, 0.0f) : 0.0f;
    S[((size_t)(b * 2048 + j)) * 2048 + m] = val;
  }
}

// ---------------------------------------------------------------------------
// Column prefix sums: Fm[b][i-1024][m] = sum_{j<i} S[b][j][m], i in [1024,2048)
// ---------------------------------------------------------------------------
__global__ __launch_bounds__(256) void prefix_kernel(const float* __restrict__ S,
                                                     float* __restrict__ Fm) {
  const int t = blockIdx.x * 256 + threadIdx.x;   // 0..4095
  const int b = t >> 11, m = t & 2047;
  float run = 0.0f;
  for (int j = 0; j < 2047; ++j) {
    run += S[((size_t)(b * 2048 + j)) * 2048 + m];
    if (j >= 1023) Fm[((size_t)(b * 1024 + (j - 1023))) * 2048 + m] = run;
  }
}

// ---------------------------------------------------------------------------
// Sequential heavy-hitter eviction scan. One block of 1024 thr per batch.
// unmasked set kept sorted ascending in LDS; argmax tie-break = first (min m).
// ---------------------------------------------------------------------------
__global__ __launch_bounds__(1024) void evict_kernel(const float* __restrict__ Fm,
                                                     int* __restrict__ evict_time) {
  __shared__ int um[1024];
  __shared__ float rs[1024];
  __shared__ int rm[1024];
  __shared__ int s_pidx, s_em;
  const int b = blockIdx.x;
  const int t = threadIdx.x;
  um[t] = t;
  evict_time[b * 2048 + t] = 1 << 30;
  evict_time[b * 2048 + 1024 + t] = 1 << 30;
  __syncthreads();
  for (int i = 1024; i < 2048; ++i) {
    const float* row = Fm + ((size_t)(b * 1024 + (i - 1024))) * 2048;
    int cand = um[t];
    rs[t] = row[cand];
    rm[t] = cand;
    __syncthreads();
    for (int s = 512; s > 0; s >>= 1) {
      if (t < s) {
        float v2 = rs[t + s]; int m2 = rm[t + s];
        if (v2 > rs[t] || (v2 == rs[t] && m2 < rm[t])) { rs[t] = v2; rm[t] = m2; }
      }
      __syncthreads();
    }
    if (t == 0) {
      int em = rm[0];
      if (row[i] > rs[0]) em = i;    // candidate i is last position: strict win only
      s_em = em;
      s_pidx = 1024;
      evict_time[b * 2048 + em] = i;
    }
    __syncthreads();
    const int em = s_em;
    if (um[t] == em) s_pidx = t;     // unique (set elements distinct)
    __syncthreads();
    const int p = s_pidx;
    int newv = um[t];
    if (p < 1024 && t >= p) newv = (t == 1023) ? i : um[t + 1];
    __syncthreads();
    um[t] = newv;
    __syncthreads();
  }
}

// ---------------------------------------------------------------------------
// Flash-style attention with WMMA, bf16 operands loaded as packed b128s.
// One wave per 16-query tile of one (b,h). Block: 128 thr (4 waves).
// Writes O in (b, n, h*64+dh) layout (ready for W_o GEMM).
// ---------------------------------------------------------------------------
__global__ __launch_bounds__(128) void attn_kernel(const unsigned short* __restrict__ Q16,
                                                   const unsigned short* __restrict__ K16,
                                                   const unsigned short* __restrict__ Vt,
                                                   const int* __restrict__ et,
                                                   float* __restrict__ O) {
  __shared__ __align__(16) unsigned short Plds[4][16][32];
  const int lane = threadIdx.x & 31;
  const int wave = threadIdx.x >> 5;
  const int tile = blockIdx.x * 4 + wave;   // 4096 tiles = 2*16*128
  const int qt = tile & 127;
  const int h = (tile >> 7) & 15;
  const int b = tile >> 11;
  const int q_base = qt * 16;
  const int half = lane >> 4, l16 = lane & 15;

  const unsigned short* Qh = Q16 + (size_t)b * 2048 * 1024 + h * 64;
  const unsigned short* Kh = K16 + (size_t)b * 2048 * 1024 + h * 64;
  const unsigned short* Vh = Vt + ((size_t)(b * 16 + h) * 64) * 2048;
  const int* etp = et + b * 2048;

  // Q as two A-fragments (16x32): elems 0..15 contiguous at K=32kk+16*half
  v16bf qf[2];
#pragma unroll
  for (int kk = 0; kk < 2; ++kk) {
    const unsigned short* qp = Qh + (size_t)(q_base + l16) * 1024 + 32 * kk + 16 * half;
    v4u lo = *(const v4u*)qp;
    v4u hi = *(const v4u*)(qp + 8);
    qf[kk] = asbf8(__builtin_shufflevector(lo, hi, 0, 1, 2, 3, 4, 5, 6, 7));
  }

  v8f acc[4];
#pragma unroll
  for (int c = 0; c < 4; ++c)
#pragma unroll
    for (int r = 0; r < 8; ++r) acc[c][r] = 0.0f;
  float m_run[8], l_run[8];
#pragma unroll
  for (int r = 0; r < 8; ++r) { m_run[r] = -1e30f; l_run[r] = 0.0f; }

  const int ktmax = (q_base + 15) >> 5;     // causal bound, 32 keys/iter
  for (int kt = 0; kt <= ktmax; ++kt) {
    const int kb = kt * 32;
    if (kt < ktmax) {   // prefetch next K/V tiles (global_prefetch_b8)
      __builtin_prefetch(Kh + (size_t)(kb + 32 + lane) * 1024, 0, 3);
      __builtin_prefetch(Vh + (size_t)(2 * lane) * 2048 + kb + 32, 0, 3);
    }
    // ---- S = Q K^T (two 16-key subtiles) ----
    v8f sacc[2];
#pragma unroll
    for (int sub = 0; sub < 2; ++sub) {
#pragma unroll
      for (int r = 0; r < 8; ++r) sacc[sub][r] = 0.0f;
#pragma unroll
      for (int kk = 0; kk < 2; ++kk) {
        // B[k][n] = K[key n][k]; elems contiguous from K = 32kk + 16*half
        v8u kv = *(const v8u*)(Kh + (size_t)(kb + 16 * sub + l16) * 1024 +
                               32 * kk + 16 * half);
        sacc[sub] = __builtin_amdgcn_wmma_f32_16x16x32_bf16(
            false, qf[kk], false, asbf8(kv), (short)0, sacc[sub], false, false);
      }
    }
    // ---- scale + causal/prune mask ----
    float sv[2][8];
#pragma unroll
    for (int sub = 0; sub < 2; ++sub) {
      const int m = kb + 16 * sub + l16;
      const int ev = etp[m];
#pragma unroll
      for (int r = 0; r < 8; ++r) {
        const int qn = q_base + r + 8 * half;
        const bool keep = (m <= qn) && (qn < ev);
        sv[sub][r] = keep ? (sacc[sub][r] * 0.125f) : -1e30f;
      }
    }
    // ---- online softmax: 16-lane butterfly row reductions ----
    float vmax[8];
#pragma unroll
    for (int r = 0; r < 8; ++r) vmax[r] = fmaxf(sv[0][r], sv[1][r]);
#pragma unroll
    for (int msk = 1; msk < 16; msk <<= 1)
#pragma unroll
      for (int r = 0; r < 8; ++r) vmax[r] = fmaxf(vmax[r], __shfl_xor(vmax[r], msk, 32));
    float corr[8];
#pragma unroll
    for (int r = 0; r < 8; ++r) {
      float nm = fmaxf(m_run[r], vmax[r]);
      corr[r] = __expf(m_run[r] - nm);
      m_run[r] = nm;
    }
    float p[2][8], vsum[8];
#pragma unroll
    for (int r = 0; r < 8; ++r) vsum[r] = 0.0f;
#pragma unroll
    for (int sub = 0; sub < 2; ++sub)
#pragma unroll
      for (int r = 0; r < 8; ++r) {
        p[sub][r] = __expf(sv[sub][r] - m_run[r]);
        vsum[r] += p[sub][r];
      }
#pragma unroll
    for (int msk = 1; msk < 16; msk <<= 1)
#pragma unroll
      for (int r = 0; r < 8; ++r) vsum[r] += __shfl_xor(vsum[r], msk, 32);
#pragma unroll
    for (int r = 0; r < 8; ++r) l_run[r] = l_run[r] * corr[r] + vsum[r];
#pragma unroll
    for (int c = 0; c < 4; ++c)
#pragma unroll
      for (int r = 0; r < 8; ++r) acc[c][r] *= corr[r];
    // ---- P: D-layout -> A-layout via per-wave LDS ----
#pragma unroll
    for (int sub = 0; sub < 2; ++sub)
#pragma unroll
      for (int r = 0; r < 8; ++r)
        Plds[wave][r + 8 * half][16 * sub + l16] = f2bu(p[sub][r]);
    asm volatile("s_wait_dscnt 0" ::: "memory");   // order wave-internal LDS exchange
    const unsigned short* pr = &Plds[wave][l16][16 * half];
    v4u plo = *(const v4u*)pr;
    v4u phi = *(const v4u*)(pr + 8);
    v16bf pf = asbf8(__builtin_shufflevector(plo, phi, 0, 1, 2, 3, 4, 5, 6, 7));
    // ---- acc += P @ V ----
#pragma unroll
    for (int c = 0; c < 4; ++c) {
      // B[key][dh]: Vt row = dh, elems contiguous from key = kb + 16*half
      v8u vv = *(const v8u*)(Vh + (size_t)(16 * c + l16) * 2048 + kb + 16 * half);
      acc[c] = __builtin_amdgcn_wmma_f32_16x16x32_bf16(
          false, pf, false, asbf8(vv), (short)0, acc[c], false, false);
    }
  }
  // ---- epilogue ----
#pragma unroll
  for (int c = 0; c < 4; ++c)
#pragma unroll
    for (int r = 0; r < 8; ++r) {
      int row = q_base + r + 8 * half;
      int col = h * 64 + 16 * c + l16;
      O[((size_t)(b * 2048 + row)) * 1024 + col] = acc[c][r] / fmaxf(l_run[r], 1e-30f);
    }
}

// ---------------------------------------------------------------------------
extern "C" void kernel_launch(void* const* d_in, const int* in_sizes, int n_in,
                              void* d_out, int out_size, void* d_ws, size_t ws_size,
                              hipStream_t stream) {
  (void)in_sizes; (void)n_in; (void)out_size; (void)ws_size;
  const float* X  = (const float*)d_in[0];
  const float* Wq = (const float*)d_in[1];
  const float* Wk = (const float*)d_in[2];
  const float* Wv = (const float*)d_in[3];
  const float* Wo = (const float*)d_in[4];
  const float* gq = (const float*)d_in[5];
  const float* bq = (const float*)d_in[6];
  const float* gk = (const float*)d_in[7];
  const float* bk = (const float*)d_in[8];
  // d_in[9..11] (cache_k, cache_v, start_pos) unused: start_pos == 0.

  // Workspace layout (bytes):
  //   Q 16MB | K 16MB | V 16MB | S 32MB (reused for O) | Fm 16MB
  //   | Q16 8MB | K16 8MB | Vt 8MB | evict_time 16KB      (peak ~120MB)
  char* w = (char*)d_ws;
  float* Qb = (float*)(w);
  float* Kb = (float*)(w + (size_t)16 * 1024 * 1024);
  float* Vb = (float*)(w + (size_t)32 * 1024 * 1024);
  float* Sb = (float*)(w + (size_t)48 * 1024 * 1024);
  float* Ob = Sb;  // S dead after prefix_kernel
  float* Fm = (float*)(w + (size_t)80 * 1024 * 1024);
  unsigned short* Q16 = (unsigned short*)(w + (size_t)96 * 1024 * 1024);
  unsigned short* K16 = (unsigned short*)(w + (size_t)104 * 1024 * 1024);
  unsigned short* Vt  = (unsigned short*)(w + (size_t)112 * 1024 * 1024);
  int* et = (int*)(w + (size_t)120 * 1024 * 1024);

  const dim3 gg(64, 8), gb(128);
  gemm_bf16_wmma<<<gg, gb, 0, stream>>>(X, Wq, Qb);
  gemm_bf16_wmma<<<gg, gb, 0, stream>>>(X, Wk, Kb);
  gemm_bf16_wmma<<<gg, gb, 0, stream>>>(X, Wv, Vb);
  ln_kernel<<<4096, 256, 0, stream>>>(Qb, gq, bq, Q16);
  ln_kernel<<<4096, 256, 0, stream>>>(Kb, gk, bk, K16);
  vtrans_kernel<<<dim3(32, 32), 256, 0, stream>>>(Vb, Vt);
  s0_kernel<<<dim3(32, 64, 2), 256, 0, stream>>>(Qb, Kb, Sb);
  prefix_kernel<<<16, 256, 0, stream>>>(Sb, Fm);
  evict_kernel<<<2, 1024, 0, stream>>>(Fm, et);
  attn_kernel<<<1024, 128, 0, stream>>>(Q16, K16, Vt, et, Ob);
  gemm_bf16_wmma<<<gg, gb, 0, stream>>>(Ob, Wo, (float*)d_out);
}
